// IIR_real_32615981646405
// MI455X (gfx1250) — matrix operand order repaired and checked by reading, not compiled, not run on hardware
//
#include <hip/hip_runtime.h>
#include <hip/hip_bf16.h>

#define FL       2048
#define SEQ      262144
#define BATCH    32
#define CDIM     16
#define NC       4095           // 2*FL-1
#define NQ       256            // SEQ / (FL/2)
#define NR       1024           // FL/2
#define XE_PAD   3071           // (FL-1) + FL/2 leading zeros
#define XE_LEN   265216         // XE_PAD + SEQ + 1  rounded to 256 multiple (1036*256)
#define GTAB     2084           // 2048 + 36 slack: no wrap masking inside a fragment

typedef __attribute__((ext_vector_type(16))) _Float16 v16h;
typedef __attribute__((ext_vector_type(8)))  float    v8f;

union HalfBits { _Float16 h; unsigned short u; };
union Frag     { v16h h; unsigned int u[8]; };

static __device__ inline unsigned short f2h_bits(float f) {
    HalfBits hb; hb.h = (_Float16)f; return hb.u;
}

// ---------------------------------------------------------------------------
// 2048-point in-place complex radix-2 FFT in LDS. 256 threads, 11 stages.
// sign = -1 forward (rfft convention), +1 inverse (caller applies 1/N).
// ---------------------------------------------------------------------------
static __device__ void fft2048(float* re, float* im, int tid, float sign) {
    for (int i = tid; i < 2048; i += 256) {          // bit-reversal (11 bits)
        int j = (int)(__brev((unsigned)i) >> 21);
        if (j > i) {
            float tr = re[i]; re[i] = re[j]; re[j] = tr;
            float ti = im[i]; im[i] = im[j]; im[j] = ti;
        }
    }
    __syncthreads();
    for (int s = 1; s <= 11; ++s) {
        int half = 1 << (s - 1);
        float ang = sign * 3.14159265358979323846f / (float)half; // 2*pi/len
        for (int idx = tid; idx < 1024; idx += 256) {
            int pos = idx & (half - 1);
            int i0  = ((idx & ~(half - 1)) << 1) | pos;
            int i1  = i0 + half;
            float sn, cs;
            __sincosf(ang * (float)pos, &sn, &cs);
            float xr = re[i1], xi = im[i1];
            float tr = cs * xr - sn * xi;
            float ti = cs * xi + sn * xr;
            float ur = re[i0], ui = im[i0];
            re[i1] = ur - tr; im[i1] = ui - ti;
            re[i0] = ur + tr; im[i0] = ui + ti;
        }
        __syncthreads();
    }
}

// ---------------------------------------------------------------------------
// K1: filters = c·Wᵀ + bias ; H = rfft(b)/(rfft(a)+1e-8) ; h = irfft(H);
//     g[j] = h[(j+1024)&2047] written as fp16 bits.  One workgroup per batch.
// ---------------------------------------------------------------------------
__global__ __launch_bounds__(256)
void build_filters_fft(const float* __restrict__ c, const float* __restrict__ W,
                       const float* __restrict__ bias, unsigned short* __restrict__ g_out) {
    __shared__ float ar[2048], ai[2048], br[2048], bi[2048];
    const int b = blockIdx.x, tid = threadIdx.x;

    float cv[CDIM];
#pragma unroll
    for (int j = 0; j < CDIM; ++j) cv[j] = c[b * CDIM + j];

    for (int n = tid; n < 4096; n += 256) {
        float f = 0.f;
        if (n < NC) {
            f = bias[n];
#pragma unroll
            for (int j = 0; j < CDIM; ++j) f += cv[j] * W[n * CDIM + j];
        }
        if (n < FL) {                 // b_coeffs
            br[n] = f; bi[n] = 0.f;
        } else {                      // a_coeffs[1..2047] = filt[2048..4094]
            int j = n - (FL - 1);     // 1..2048
            if (j < FL) ar[j] = f;
            ai[n - FL] = 0.f;
        }
    }
    if (tid == 0) ar[0] = 1.0f;       // a0 = 1
    __syncthreads();

    fft2048(br, bi, tid, -1.0f);      // B = rfft(b_coeffs)
    fft2048(ar, ai, tid, -1.0f);      // A = rfft(a_coeffs)

    for (int k = tid; k < 2048; k += 256) {   // H = B/(A + 1e-8) (real eps, as JAX)
        float arr = ar[k] + 1e-8f, aii = ai[k];
        float den = arr * arr + aii * aii;
        float inv = 1.0f / den;
        float hr = (br[k] * arr + bi[k] * aii) * inv;
        float hi = (bi[k] * arr - br[k] * aii) * inv;
        br[k] = hr; bi[k] = hi;
    }
    __syncthreads();

    fft2048(br, bi, tid, 1.0f);       // h = ifft(H) (scale 1/N below)

    const float scl = 1.0f / 2048.0f;
    for (int j = tid; j < 2048; j += 256)
        g_out[b * 2048 + j] = f2h_bits(br[(j + 1024) & 2047] * scl);
}

// ---------------------------------------------------------------------------
// K2: zero-padded fp16 copy of the (batch-shared) input signal.
// ---------------------------------------------------------------------------
__global__ __launch_bounds__(256)
void build_xe(const float* __restrict__ x, unsigned short* __restrict__ xe) {
    int i = blockIdx.x * 256 + threadIdx.x;
    if (i < XE_LEN) {
        int s = i - XE_PAD;
        float v = (s >= 0 && s < SEQ) ? x[s] : 0.f;
        xe[i] = f2h_bits(v);
    }
}

// ---------------------------------------------------------------------------
// K3: out[b, q*1024 + r] = sum_m xe[q*1024+m] * g_b[(r-m)&2047]
//     GEMM  M=q (256), N=r (1024), K=m (2048) per batch via v_wmma f16.
//     Block = 64 q x 256 r tile, 8 waves: each wave 4 q-tiles x 2 r-tiles.
//
//     Circulant B fragments come from *reversed* g tables so each K-pair is
//     one aligned dword in LDS:
//       grA[i] = g[(-i)&2047]  -> pair (g[r-m], g[r-m-1]) == word at s=(m-r)&2047
//       grB[i] = grA[i+1]      -> same trick for odd-parity lanes
//     8 consecutive dwords per fragment off one base -> ds_load_2addr fusion.
// ---------------------------------------------------------------------------
__global__ __launch_bounds__(256)
void circ_gemm(const unsigned short* __restrict__ gball,
               const unsigned short* __restrict__ xe,
               float* __restrict__ out) {
    __shared__ __align__(16) unsigned short grA[GTAB];
    __shared__ __align__(16) unsigned short grB[GTAB];
    const int b  = blockIdx.z;
    const int q0 = blockIdx.x * 64;
    const int r0 = blockIdx.y * 256;
    const int tid = threadIdx.x;

    const unsigned short* gb = gball + b * 2048;
    for (int i = tid; i < GTAB; i += 256) {
        grA[i] = gb[(-i) & 2047];
        grB[i] = gb[(-(i + 1)) & 2047];
    }
    __syncthreads();

    const int w      = tid >> 5;
    const int lane   = tid & 31;
    const int lanelo = lane & 15;
    const int lanehi = lane >> 4;

    // parity of r (== parity of s) is constant per lane -> pick table once
    const unsigned* tab = (lanelo & 1) ? (const unsigned*)grB : (const unsigned*)grA;

    v8f acc[4][2];
#pragma unroll
    for (int i = 0; i < 4; ++i)
#pragma unroll
        for (int j = 0; j < 2; ++j) acc[i][j] = (v8f)(0.0f);

    const uint4* xe4 = (const uint4*)xe;                 // 16B-aligned K-pair stream

    for (int k0 = 0; k0 < 2048; k0 += 32) {
        // ---- A fragments: X[q, m] straight from global (layout-native) ----
        Frag A[4];
#pragma unroll
        for (int i = 0; i < 4; ++i) {
            int qa   = q0 + i * 16 + lanelo;
            int base = qa * 1024 + k0 + lanehi * 16;     // half index, %8 == 0
            uint4 lo = xe4[(base >> 3)];                 // K = base .. base+7
            uint4 hi = xe4[(base >> 3) + 1];             // K = base+8 .. +15
            A[i].u[0] = lo.x; A[i].u[1] = lo.y; A[i].u[2] = lo.z; A[i].u[3] = lo.w;
            A[i].u[4] = hi.x; A[i].u[5] = hi.y; A[i].u[6] = hi.z; A[i].u[7] = hi.w;
        }
        if (k0 + 32 < 2048) {                            // stream hint for next K chunk
            int qa = q0 + lanelo;
            __builtin_prefetch(xe + qa * 1024 + k0 + 32, 0, 1);
        }
        // ---- B fragments: 8 consecutive aligned dwords per fragment ----
        Frag B[2];
#pragma unroll
        for (int j = 0; j < 2; ++j) {
            int rB    = r0 + (w * 2 + j) * 16 + lanelo;
            int sBase = (k0 + lanehi * 16 - rB) & 2047;  // parity == parity(rB)
            unsigned wb = (unsigned)sBase >> 1;
#pragma unroll
            for (int v = 0; v < 8; ++v)
                B[j].u[v] = tab[wb + v];
        }
        // ---- 8 WMMAs: D = A x B + C, fp32 accumulate ----
#pragma unroll
        for (int i = 0; i < 4; ++i)
#pragma unroll
            for (int j = 0; j < 2; ++j)
                acc[i][j] = __builtin_amdgcn_wmma_f32_16x16x32_f16(
                    false, A[i].h, false, B[j].h, (short)0, acc[i][j], false, false);
    }

    // ---- epilogue: lanes 0-15 map to consecutive r -> coalesced stores ----
#pragma unroll
    for (int i = 0; i < 4; ++i)
#pragma unroll
        for (int j = 0; j < 2; ++j) {
            int rr = r0 + (w * 2 + j) * 16 + lanelo;
#pragma unroll
            for (int vc = 0; vc < 8; ++vc) {
                int qq = q0 + i * 16 + vc + lanehi * 8;
                out[(size_t)b * SEQ + (size_t)qq * 1024 + rr] = acc[i][j][vc];
            }
        }
}

// ---------------------------------------------------------------------------
extern "C" void kernel_launch(void* const* d_in, const int* in_sizes, int n_in,
                              void* d_out, int out_size, void* d_ws, size_t ws_size,
                              hipStream_t stream) {
    (void)in_sizes; (void)n_in; (void)out_size; (void)ws_size;
    const float* x    = (const float*)d_in[0];   // (1, 262144, 1)
    const float* c    = (const float*)d_in[1];   // (32, 1, 16)
    const float* W    = (const float*)d_in[2];   // (4095, 16)
    const float* bias = (const float*)d_in[3];   // (4095,)
    float* out = (float*)d_out;                  // (32, 262144, 1) f32

    unsigned short* g  = (unsigned short*)d_ws;                         // 32*2048 fp16
    unsigned short* xe = (unsigned short*)((char*)d_ws + BATCH * 2048 * sizeof(unsigned short));

    build_filters_fft<<<BATCH, 256, 0, stream>>>(c, W, bias, g);
    build_xe<<<XE_LEN / 256, 256, 0, stream>>>(x, xe);

    dim3 grid(NQ / 64, NR / 256, BATCH);         // (4, 4, 32)
    circ_gemm<<<grid, 256, 0, stream>>>(g, xe, out);
}
